// VariationalHMMGating_5686536700171
// MI455X (gfx1250) — matrix operand order repaired and checked by reading, not compiled, not run on hardware
//
#include <hip/hip_runtime.h>
#include <hip/hip_bf16.h>
#include <stdint.h>

typedef __attribute__((ext_vector_type(16))) _Float16 v16h;
typedef __attribute__((ext_vector_type(8)))  _Float16 v8h;
typedef __attribute__((ext_vector_type(8)))  float    v8f;

#define B_SZ   1024
#define S_SZ   512
#define D_SZ   16
#define H_SZ   256
#define K_SZ   8
#define G4H    1024           // 4*H
#define TOKENS ((S_SZ - 1) * B_SZ)   // 523264
#define TOK_PER_WG 4
#define ROWS   (TOK_PER_WG * K_SZ)   // 32 candidate rows per WG

// ---------------- device helpers ----------------

__device__ __forceinline__ float sLSTM_cell(float ip, float fp, float zp, float op) {
    // h0 = c0 = n0 = 0 one-step cell with exponential gating + stabilizer
    float m  = fmaxf(fp, ip);
    float i  = __expf(ip - m);
    float c  = i * tanhf(zp);
    float sg = 1.0f / (1.0f + __expf(-op));
    return sg * c / fmaxf(i, 1e-6f);
}

// A fragment (16x32 f16, M x K), wave32 layout:
// lanes 0-15: M=lane, K = {k..k+7, k+16..k+23}; lanes 16-31: M=lane-16, K = {k+8..k+15, k+24..k+31}
__device__ __forceinline__ v16h load_a_frag(const _Float16* base, int m0, int k, int lane, int ld) {
    int half = lane >> 4;
    int m    = m0 + (lane & 15);
    const _Float16* p = base + m * ld + k + half * 8;
    v8h lo = *(const v8h*)(p);
    v8h hi = *(const v8h*)(p + 16);
    v16h r;
#pragma unroll
    for (int i = 0; i < 8; ++i) { r[i] = lo[i]; r[i + 8] = hi[i]; }
    return r;
}

// B fragment (32x16 f16, K x N). Weight W is row-major [N][Kfull]; B[k][n] = W[n][k].
// lanes 0-15: N=lane, K = {k..k+7, k+16..k+23}; lanes 16-31: N=lane-16, K = {k+8..k+15, k+24..k+31}
__device__ __forceinline__ v16h load_b_frag(const _Float16* w, int n0, int k, int lane, int ld) {
    int n = n0 + (lane & 15);
    const _Float16* p = w + n * ld + k + (lane >> 4) * 8;
    v8h lo = *(const v8h*)(p);
    v8h hi = *(const v8h*)(p + 16);
    v16h r;
#pragma unroll
    for (int i = 0; i < 8; ++i) { r[i] = lo[i]; r[i + 8] = hi[i]; }
    return r;
}

__device__ __forceinline__ v8f wmma_f16(v16h a, v16h b, v8f c) {
    return __builtin_amdgcn_wmma_f32_16x16x32_f16(false, a, false, b, (short)0, c, false, false);
}

// ---------------- kernel 1: weight prep ----------------
__global__ void __launch_bounds__(256) prep_kernel(
    const float* __restrict__ tW0, const float* __restrict__ tb0,
    const float* __restrict__ tW1, const float* __restrict__ fctW,
    _Float16* __restrict__ tW1h, _Float16* __restrict__ fctWh,
    float* __restrict__ colbias)
{
    int i = blockIdx.x * 256 + threadIdx.x;
    if (i < G4H * H_SZ) tW1h[i]  = (_Float16)tW1[i];      // (1024,256)
    if (i < 64 * H_SZ)  fctWh[i] = (_Float16)fctW[i];     // (64,256)
    if (i < K_SZ * G4H) {                                  // colbias[s][j] = tW0[j][16+s] + tb0[j]
        int s = i >> 10, j = i & 1023;
        colbias[i] = tW0[j * (D_SZ + K_SZ) + D_SZ + s] + tb0[j];
    }
}

// ---------------- kernel 2: t=0 init path (tiny) ----------------
__global__ void __launch_bounds__(256) init_kernel(
    const float* __restrict__ x, const float* __restrict__ unif,
    const float* __restrict__ iW0, const float* __restrict__ ib0,
    const float* __restrict__ iW1, const float* __restrict__ ib1,
    const float* __restrict__ fciW, const float* __restrict__ fcib,
    float* __restrict__ initp, int* __restrict__ out)
{
    int b = blockIdx.x, tid = threadIdx.x;
    __shared__ float xs[D_SZ];
    __shared__ float h1[H_SZ];
    __shared__ float h2[H_SZ];
    __shared__ float lg[K_SZ];
    if (tid < D_SZ) xs[tid] = x[(b * S_SZ) * D_SZ + tid];
    __syncthreads();
    float pre[4];
#pragma unroll
    for (int g = 0; g < 4; ++g) {
        int j = g * H_SZ + tid;
        const float* w = iW0 + j * D_SZ;
        float acc = ib0[j];
#pragma unroll
        for (int d = 0; d < D_SZ; ++d) acc += xs[d] * w[d];
        pre[g] = acc;
    }
    h1[tid] = sLSTM_cell(pre[0], pre[1], pre[2], pre[3]);
    __syncthreads();
#pragma unroll
    for (int g = 0; g < 4; ++g) {
        int j = g * H_SZ + tid;
        const float* w = iW1 + j * H_SZ;
        float acc = ib1[j];
        for (int i = 0; i < H_SZ; ++i) acc += h1[i] * w[i];
        pre[g] = acc;
    }
    h2[tid] = sLSTM_cell(pre[0], pre[1], pre[2], pre[3]);
    __syncthreads();
    if (tid < K_SZ) {
        const float* w = fciW + tid * H_SZ;
        float acc = fcib[tid];
        for (int i = 0; i < H_SZ; ++i) acc += h2[i] * w[i];
        lg[tid] = acc;
    }
    __syncthreads();
    if (tid == 0) {
        float mx = lg[0];
#pragma unroll
        for (int k = 1; k < K_SZ; ++k) mx = fmaxf(mx, lg[k]);
        float e[K_SZ], sum = 0.f;
#pragma unroll
        for (int k = 0; k < K_SZ; ++k) { e[k] = __expf(lg[k] - mx); sum += e[k]; }
        float inv = 1.0f / sum;
        int best = 0; float bv = -1e30f;
        const float* u = unif + (b * S_SZ) * K_SZ;
#pragma unroll
        for (int k = 0; k < K_SZ; ++k) {
            float p = e[k] * inv;
            initp[b * K_SZ + k] = p;
            float gum = -logf(-logf(u[k] + 1e-12f) + 1e-12f);
            float v = logf(p + 1e-12f) + gum;
            if (v > bv) { bv = v; best = k; }
        }
        out[b * S_SZ + 0] = best;
    }
}

// ---------------- kernel 3: fused candidate pipeline (WMMA workhorse) ----------------
__global__ void __launch_bounds__(256) fused_kernel(
    const float* __restrict__ x, const float* __restrict__ unif,
    const float* __restrict__ tW0, const float* __restrict__ tb1,
    const float* __restrict__ fctb,
    const _Float16* __restrict__ tW1h, const _Float16* __restrict__ fctWh,
    const float* __restrict__ colbias, const float* __restrict__ initp,
    unsigned char* __restrict__ tau)
{
    __shared__ union { float xw[TOK_PER_WG * G4H]; _Float16 h2[ROWS * H_SZ]; } r1; // 16 KB
    __shared__ _Float16 h1s[ROWS * H_SZ];    // 16 KB
    __shared__ float    lgs[ROWS * 64];      // 8 KB
    __shared__ float    xt[TOK_PER_WG * D_SZ];

    const int tid  = threadIdx.x;
    const int wave = tid >> 5;
    const int lane = tid & 31;
    const int tok0 = blockIdx.x * TOK_PER_WG;

    // ---- async-load the x tile (4 tokens x 16 f32) straight into LDS (ASYNCcnt path) ----
    if (tid < TOK_PER_WG * D_SZ) {   // waves 0 and 1, full EXEC
        int tk = tid >> 4, d = tid & 15;
        int tok = tok0 + tk; if (tok >= TOKENS) tok = TOKENS - 1;
        int b = tok / (S_SZ - 1), t = 1 + tok % (S_SZ - 1);
        unsigned lds_off = (unsigned)(uintptr_t)(&xt[tk * D_SZ + d]);   // low 32 bits = LDS offset
        unsigned long long ga = (unsigned long long)(uintptr_t)(x + (b * S_SZ + t) * D_SZ + d);
        asm volatile("global_load_async_to_lds_b32 %0, %1, off"
                     :: "v"(lds_off), "v"(ga) : "memory");
    }
    asm volatile("s_wait_asynccnt 0x0" ::: "memory");
    __syncthreads();

    // ---- step 1a: xw[tk][j] = x_t . tW0[j][0:16]  (shared across the 8 candidates) ----
    for (int o = tid; o < TOK_PER_WG * G4H; o += 256) {
        int tk = o >> 10, j = o & 1023;
        const float* w = tW0 + j * (D_SZ + K_SZ);
        float acc = 0.f;
#pragma unroll
        for (int d = 0; d < D_SZ; ++d) acc += xt[tk * D_SZ + d] * w[d];
        r1.xw[o] = acc;
    }
    __syncthreads();

    // ---- step 1b: layer-0 cell -> h1 (32 x 256 f16, the WMMA A tile) ----
    for (int o = tid; o < ROWS * H_SZ; o += 256) {
        int r = o >> 8, i = o & 255;
        int tk = r >> 3, s = r & 7;
        const float* cb = colbias + s * G4H;
        const float* xw = r1.xw + tk * G4H;
        float ip = xw[i]           + cb[i];
        float fp = xw[H_SZ + i]    + cb[H_SZ + i];
        float zp = xw[2*H_SZ + i]  + cb[2*H_SZ + i];
        float op = xw[3*H_SZ + i]  + cb[3*H_SZ + i];
        h1s[o] = (_Float16)sLSTM_cell(ip, fp, zp, op);
    }
    __syncthreads();

    // ---- step 2: layer-1 GEMM (32x256)@(256x1024) via WMMA.
    //      Register-block BOTH 16-row M-tiles per task so every B fragment feeds
    //      two WMMAs (32 FLOP/B from L2 instead of 16); 4-gate grouping keeps the
    //      full pre-activation for a 16x16 hidden tile in registers. ----
    for (int hc = wave; hc < H_SZ / 16; hc += 8) {   // 16 hidden chunks / 8 waves = 2 each
        v8f acc[2][4];
#pragma unroll
        for (int mt = 0; mt < 2; ++mt)
#pragma unroll
            for (int g = 0; g < 4; ++g) acc[mt][g] = (v8f){};
        for (int k = 0; k < H_SZ; k += 32) {
            v16h a0 = load_a_frag(h1s,  0, k, lane, H_SZ);
            v16h a1 = load_a_frag(h1s, 16, k, lane, H_SZ);
            __builtin_prefetch(tW1h + (hc * 16) * H_SZ + k + 64, 0, 1);
#pragma unroll
            for (int g = 0; g < 4; ++g) {
                v16h bfr = load_b_frag(tW1h, g * H_SZ + hc * 16, k, lane, H_SZ);
                acc[0][g] = wmma_f16(a0, bfr, acc[0][g]);
                acc[1][g] = wmma_f16(a1, bfr, acc[1][g]);
            }
        }
        // bias + cell, entirely in registers; C layout: lane->N, vgpr e->M
        int col = hc * 16 + (lane & 15);
        float bi = tb1[col], bf = tb1[H_SZ + col], bz = tb1[2*H_SZ + col], bo = tb1[3*H_SZ + col];
#pragma unroll
        for (int mt = 0; mt < 2; ++mt) {
            int rbase = mt * 16 + (lane >> 4) * 8;
#pragma unroll
            for (int e = 0; e < 8; ++e) {
                float h = sLSTM_cell(acc[mt][0][e] + bi, acc[mt][1][e] + bf,
                                     acc[mt][2][e] + bz, acc[mt][3][e] + bo);
                r1.h2[(rbase + e) * H_SZ + col] = (_Float16)h;
            }
        }
    }
    __syncthreads();

    // ---- step 3: fct GEMM (32x256)@(256x64) via WMMA -> logits ----
    for (int task = wave; task < 2 * 4; task += 8) {   // 2 M-tiles x 4 N-tiles
        int mtile = task >> 2, nt = task & 3;
        v8f acc = {};
        for (int k = 0; k < H_SZ; k += 32) {
            v16h af = load_a_frag(r1.h2, mtile * 16, k, lane, H_SZ);
            v16h bf = load_b_frag(fctWh, nt * 16, k, lane, H_SZ);
            acc = wmma_f16(af, bf, acc);
        }
        int col = nt * 16 + (lane & 15);
        float bias = fctb[col];
        int rbase = mtile * 16 + (lane >> 4) * 8;
#pragma unroll
        for (int e = 0; e < 8; ++e) lgs[(rbase + e) * 64 + col] = acc[e] + bias;
    }
    __syncthreads();

    // ---- step 4: per candidate row: softmax(T rows), nprob = initp . T, gumbel argmax ----
    if (tid < ROWS) {
        int r = tid, tk = r >> 3, s = r & 7;
        int tok = tok0 + tk;
        if (tok < TOKENS) {
            int b = tok / (S_SZ - 1), t = 1 + tok % (S_SZ - 1);
            const float* l = lgs + r * 64;
            float mx[K_SZ], sm[K_SZ];
#pragma unroll
            for (int kk = 0; kk < K_SZ; ++kk) {
                float m = l[kk * 8];
#pragma unroll
                for (int j = 1; j < K_SZ; ++j) m = fmaxf(m, l[kk * 8 + j]);
                float sum = 0.f;
#pragma unroll
                for (int j = 0; j < K_SZ; ++j) sum += __expf(l[kk * 8 + j] - m);
                mx[kk] = m; sm[kk] = sum;
            }
            const float* p = initp + b * K_SZ;
            float np[K_SZ];
#pragma unroll
            for (int j = 0; j < K_SZ; ++j) np[j] = 0.f;
#pragma unroll
            for (int kk = 0; kk < K_SZ; ++kk) {
                float scale = p[kk] / sm[kk];
#pragma unroll
                for (int j = 0; j < K_SZ; ++j) np[j] += scale * __expf(l[kk * 8 + j] - mx[kk]);
            }
            const float* u = unif + (b * S_SZ + t) * K_SZ;
            int best = 0; float bv = -1e30f;
#pragma unroll
            for (int j = 0; j < K_SZ; ++j) {
                float gum = -logf(-logf(u[j] + 1e-12f) + 1e-12f);
                float v = logf(np[j] + 1e-12f) + gum;
                if (v > bv) { bv = v; best = j; }
            }
            tau[(b * S_SZ + t) * K_SZ + s] = (unsigned char)best;
        }
    }
}

// ---------------- kernel 4: trivial lookup-chain scan ----------------
__global__ void __launch_bounds__(256) scan_kernel(
    const unsigned char* __restrict__ tau, int* __restrict__ out)
{
    int b = blockIdx.x * blockDim.x + threadIdx.x;
    if (b >= B_SZ) return;
    int s = out[b * S_SZ];                    // s0 from init_kernel
    const unsigned char* row = tau + b * S_SZ * K_SZ;
    for (int t = 1; t < S_SZ; ++t) {
        s = (int)row[t * K_SZ + s];
        out[b * S_SZ + t] = s;
    }
}

// ---------------- host launcher ----------------
extern "C" void kernel_launch(void* const* d_in, const int* in_sizes, int n_in,
                              void* d_out, int out_size, void* d_ws, size_t ws_size,
                              hipStream_t stream) {
    (void)in_sizes; (void)n_in; (void)out_size; (void)ws_size;
    const float* x    = (const float*)d_in[0];
    const float* unif = (const float*)d_in[1];
    const float* iW0  = (const float*)d_in[2];
    const float* ib0  = (const float*)d_in[4];
    const float* iW1  = (const float*)d_in[5];
    const float* ib1  = (const float*)d_in[7];
    const float* tW0  = (const float*)d_in[8];
    const float* tb0  = (const float*)d_in[10];
    const float* tW1  = (const float*)d_in[11];
    const float* tb1  = (const float*)d_in[13];
    const float* fciW = (const float*)d_in[14];
    const float* fcib = (const float*)d_in[15];
    const float* fctW = (const float*)d_in[16];
    const float* fctb = (const float*)d_in[17];
    int* out = (int*)d_out;

    // workspace carve-up (~4.6 MB total)
    char* ws = (char*)d_ws;
    float*          colbias = (float*)(ws);                 // 8*1024 f32   = 32 KB
    float*          initp   = (float*)(ws + (32u << 10));   // 1024*8 f32   = 32 KB
    _Float16*       tW1h    = (_Float16*)(ws + (64u << 10));// 1024*256 f16 = 512 KB
    _Float16*       fctWh   = (_Float16*)(ws + (576u << 10));// 64*256 f16  = 32 KB
    unsigned char*  tau     = (unsigned char*)(ws + (608u << 10)); // 1024*512*8 = 4 MB

    prep_kernel<<<(G4H * H_SZ + 255) / 256, 256, 0, stream>>>(
        tW0, tb0, tW1, fctW, tW1h, fctWh, colbias);

    init_kernel<<<B_SZ, 256, 0, stream>>>(
        x, unif, iW0, ib0, iW1, ib1, fciW, fcib, initp, out);

    fused_kernel<<<TOKENS / TOK_PER_WG, 256, 0, stream>>>(
        x, unif, tW0, tb1, fctb, tW1h, fctWh, colbias, initp, tau);

    scan_kernel<<<(B_SZ + 255) / 256, 256, 0, stream>>>(tau, out);
}
